// GCN_moe_13675175871112
// MI455X (gfx1250) — matrix operand order repaired and checked by reading, not compiled, not run on hardware
//
#include <hip/hip_runtime.h>
#include <hip/hip_bf16.h>

#define NN     50000
#define DD     512
#define NE_    4
#define NEDGE  400000
#define EPS_   1e-5f

typedef __bf16 v16bf __attribute__((ext_vector_type(16)));
typedef __bf16 v8bf  __attribute__((ext_vector_type(8)));
typedef float  v8f   __attribute__((ext_vector_type(8)));

__device__ __forceinline__ unsigned short f2bf(float f) {
    unsigned u = __float_as_uint(f);
    unsigned r = u + 0x7FFFu + ((u >> 16) & 1u);   // round-to-nearest-even
    return (unsigned short)(r >> 16);
}

// ---------------- degree / dinv (shared by both layers) ----------------
__global__ void k_set1(float* deg) {
    int i = blockIdx.x * blockDim.x + threadIdx.x;
    if (i < NN) deg[i] = 1.0f;   // self loop
}
__global__ void k_deg_edges(const int* __restrict__ ei, float* __restrict__ deg) {
    int e = blockIdx.x * blockDim.x + threadIdx.x;
    if (e < NEDGE) atomicAdd(&deg[ei[NEDGE + e]], 1.0f);
}
__global__ void k_rsqrt(const float* __restrict__ deg, float* __restrict__ dinv) {
    int i = blockIdx.x * blockDim.x + threadIdx.x;
    if (i < NN) dinv[i] = rsqrtf(deg[i]);
}

// ---------------- per-layer small init ----------------
__global__ void k_zero_small(float* colsum, float* colsq, int* counts) {
    int i = blockIdx.x * blockDim.x + threadIdx.x;
    if (i < DD) { colsum[i] = 0.f; colsq[i] = 0.f; }
    if (i < NE_) counts[i] = 0;
}

// ---------------- GraphNorm: column sums + finalize + apply ----------------
__global__ void k_colstats(const float* __restrict__ x,
                           float* __restrict__ s, float* __restrict__ sq) {
    int d0 = threadIdx.x;                 // 0..255
    float a0 = 0.f, q0 = 0.f, a1 = 0.f, q1 = 0.f;
    for (int r = blockIdx.x; r < NN; r += gridDim.x) {
        const float* row = x + (size_t)r * DD;
        float v0 = row[d0], v1 = row[d0 + 256];
        a0 += v0; q0 += v0 * v0; a1 += v1; q1 += v1 * v1;
    }
    atomicAdd(&s[d0], a0);        atomicAdd(&sq[d0], q0);
    atomicAdd(&s[d0 + 256], a1);  atomicAdd(&sq[d0 + 256], q1);
}
__global__ void k_gn_finalize(const float* __restrict__ s, const float* __restrict__ sq,
                              const float* __restrict__ gw, const float* __restrict__ gb,
                              const float* __restrict__ gms,
                              float* __restrict__ scalev, float* __restrict__ shiftv) {
    int d = threadIdx.x;
    if (d < DD) {
        float mu  = s[d] * (1.0f / NN);
        float ex2 = sq[d] * (1.0f / NN);
        float ms  = gms[d];
        float var = ex2 - mu * mu * ms * (2.0f - ms);   // E[(x-ms*mu)^2]
        float sc  = gw[d] / sqrtf(var + EPS_);
        scalev[d] = sc;
        shiftv[d] = gb[d] - sc * ms * mu;
    }
}
__global__ void k_gn_apply(const float* __restrict__ x,
                           const float* __restrict__ scalev, const float* __restrict__ shiftv,
                           float* __restrict__ h) {
    size_t i = (size_t)blockIdx.x * blockDim.x + threadIdx.x;
    if (i < (size_t)NN * DD) {
        int d = (int)(i & (DD - 1));
        h[i] = x[i] * scalev[d] + shiftv[d];
    }
}

// ---------------- gating: logits = h @ wg, argmax; one wave per node ----------------
__global__ __launch_bounds__(256) void k_gate(const float* __restrict__ h,
                                              const float* __restrict__ wg,   // [D, NE]
                                              int* __restrict__ eid, int* __restrict__ counts) {
    __shared__ float wgs[DD * NE_];
    for (int i = threadIdx.x; i < DD * NE_; i += 256) wgs[i] = wg[i];
    __syncthreads();
    int wave = threadIdx.x >> 5;
    int lane = threadIdx.x & 31;
    int node = blockIdx.x * 8 + wave;
    if (node >= NN) return;
    float acc[NE_] = {0.f, 0.f, 0.f, 0.f};
    const float* row = h + (size_t)node * DD;
    for (int k = lane; k < DD; k += 32) {
        float xv = row[k];
        #pragma unroll
        for (int e = 0; e < NE_; ++e) acc[e] += xv * wgs[k * NE_ + e];
    }
    #pragma unroll
    for (int e = 0; e < NE_; ++e) {
        float v = acc[e];
        for (int off = 16; off > 0; off >>= 1) v += __shfl_xor(v, off, 32);
        acc[e] = v;
    }
    if (lane == 0) {
        int be = 0; float bv = acc[0];
        #pragma unroll
        for (int e = 1; e < NE_; ++e) if (acc[e] > bv) { bv = acc[e]; be = e; }
        eid[node] = be;
        atomicAdd(&counts[be], 1);
    }
}
__global__ void k_offsets(const int* __restrict__ counts, int* __restrict__ offs,
                          int* __restrict__ cursor) {
    if (threadIdx.x == 0 && blockIdx.x == 0) {
        int o = 0;
        for (int e = 0; e < NE_; ++e) { offs[e] = o; cursor[e] = o; o += counts[e]; }
    }
}
__global__ void k_scatter(const int* __restrict__ eid, int* __restrict__ cursor,
                          int* __restrict__ perm) {
    int n = blockIdx.x * blockDim.x + threadIdx.x;
    if (n < NN) {
        int pos = atomicAdd(&cursor[eid[n]], 1);
        perm[pos] = n;
    }
}

// ---------------- GCN propagate: self loop init + edge scatter-add ----------------
__global__ void k_prop_init(const float* __restrict__ h, const float* __restrict__ dinv,
                            float* __restrict__ agg) {
    size_t i = (size_t)blockIdx.x * blockDim.x + threadIdx.x;
    if (i < (size_t)NN * DD) {
        int n = (int)(i >> 9);
        float di = dinv[n];
        agg[i] = h[i] * di * di;
    }
}
__global__ __launch_bounds__(256) void k_prop_edges(const float* __restrict__ h,
                                                    const int* __restrict__ ei,
                                                    const float* __restrict__ dinv,
                                                    float* __restrict__ agg) {
    int sub  = threadIdx.x & 63;                    // 64 threads per edge, 8 ch each
    int eIdx = blockIdx.x * 4 + (threadIdx.x >> 6);
    if (eIdx >= NEDGE) return;
    int s = ei[eIdx], t = ei[NEDGE + eIdx];
    float w = dinv[s] * dinv[t];
    const float4* hp = (const float4*)(h + (size_t)s * DD + sub * 8);
    float4 v0 = hp[0], v1 = hp[1];
    float* ap = agg + (size_t)t * DD + sub * 8;
    atomicAdd(ap + 0, v0.x * w); atomicAdd(ap + 1, v0.y * w);
    atomicAdd(ap + 2, v0.z * w); atomicAdd(ap + 3, v0.w * w);
    atomicAdd(ap + 4, v1.x * w); atomicAdd(ap + 5, v1.y * w);
    atomicAdd(ap + 6, v1.z * w); atomicAdd(ap + 7, v1.w * w);
}

// ---------------- conversions ----------------
__global__ void k_convertA(const float* __restrict__ agg, unsigned short* __restrict__ Abf) {
    size_t i = (size_t)blockIdx.x * blockDim.x + threadIdx.x;
    if (i < (size_t)NN * DD) Abf[i] = f2bf(agg[i]);
}
__global__ void k_convert_Wt(const float* __restrict__ W, unsigned short* __restrict__ Wt) {
    int i = blockIdx.x * blockDim.x + threadIdx.x;     // NE*D*D = 1,048,576
    if (i < NE_ * DD * DD) {
        int e = i >> 18, rem = i & 262143;
        int k = rem >> 9, n = rem & 511;
        Wt[(size_t)e * DD * DD + (size_t)n * DD + k] = f2bf(W[i]);
    }
}

// ------- grouped-by-expert WMMA GEMM, 32x32 C tile per wave (2x2 register blocking) -------
__global__ __launch_bounds__(32) void k_gemm_moe(const unsigned short* __restrict__ Abf,
                                                 const unsigned short* __restrict__ Wt,
                                                 const float* __restrict__ bias,   // [NE,D]
                                                 const int* __restrict__ perm,
                                                 const int* __restrict__ offs,
                                                 const int* __restrict__ counts,
                                                 float* __restrict__ out0,
                                                 float* __restrict__ out1) {
    const int e     = blockIdx.z;
    const int cnt   = counts[e];
    const int mbase = blockIdx.x * 32;
    if (mbase >= cnt) return;                 // uniform exit: EXEC stays all-ones
    const int lane    = threadIdx.x;          // 0..31
    const int sub     = lane & 15;
    const int colbase = blockIdx.y * 32;
    const int off     = offs[e];

    // A fragments: this lane supplies rows M = sub and 16+sub of the 32-row tile
    int at0 = mbase + sub;
    int at1 = mbase + 16 + sub;
    int arow0 = perm[off + (at0 < cnt ? at0 : cnt - 1)];
    int arow1 = perm[off + (at1 < cnt ? at1 : cnt - 1)];
    const unsigned short* Ap0 = Abf + (size_t)arow0 * DD;
    const unsigned short* Ap1 = Abf + (size_t)arow1 * DD;

    const int hb   = (lane < 16) ? 0 : 8;     // ISA 16-bit A 16x32 layout
    const int kb16 = (lane < 16) ? 0 : 16;    // unused; B is loaded with hb, same k order as A
    const int ncol0 = colbase + sub;
    const int ncol1 = colbase + 16 + sub;
    const unsigned short* Bp0 = Wt + (size_t)e * DD * DD + (size_t)ncol0 * DD;
    const unsigned short* Bp1 = Wt + (size_t)e * DD * DD + (size_t)ncol1 * DD;

    v8f c00 = {}, c01 = {}, c10 = {}, c11 = {};
    for (int k0 = 0; k0 < DD; k0 += 32) {
        v8bf a0l = *(const v8bf*)(Ap0 + k0 + hb);
        v8bf a0h = *(const v8bf*)(Ap0 + k0 + 16 + hb);
        v8bf a1l = *(const v8bf*)(Ap1 + k0 + hb);
        v8bf a1h = *(const v8bf*)(Ap1 + k0 + 16 + hb);
        v8bf b0l = *(const v8bf*)(Bp0 + k0 + hb);
        v8bf b0h = *(const v8bf*)(Bp0 + k0 + 16 + hb);
        v8bf b1l = *(const v8bf*)(Bp1 + k0 + hb);
        v8bf b1h = *(const v8bf*)(Bp1 + k0 + 16 + hb);
        v16bf a0, a1, b0, b1;
        #pragma unroll
        for (int i = 0; i < 8; ++i) {
            a0[i] = a0l[i]; a0[8 + i] = a0h[i];
            a1[i] = a1l[i]; a1[8 + i] = a1h[i];
            b0[i] = b0l[i]; b0[8 + i] = b0h[i];
            b1[i] = b1l[i]; b1[8 + i] = b1h[i];
        }
        c00 = __builtin_amdgcn_wmma_f32_16x16x32_bf16(false, a0, false, b0, (short)0, c00, false, false);
        c01 = __builtin_amdgcn_wmma_f32_16x16x32_bf16(false, a0, false, b1, (short)0, c01, false, false);
        c10 = __builtin_amdgcn_wmma_f32_16x16x32_bf16(false, a1, false, b0, (short)0, c10, false, false);
        c11 = __builtin_amdgcn_wmma_f32_16x16x32_bf16(false, a1, false, b1, (short)0, c11, false, false);
    }

    // C layout per 16x16 subtile: VGPR r, lane: M = r + (lane<16?0:8), N = lane%16
    const int mh  = (lane < 16) ? 0 : 8;
    const float bv0 = bias[e * DD + ncol0];
    const float bv1 = bias[e * DD + ncol1];
    #pragma unroll
    for (int r = 0; r < 8; ++r) {
        int mt0 = mbase + r + mh;          // M subtile 0
        int mt1 = mbase + 16 + r + mh;     // M subtile 1
        if (mt0 < cnt) {
            int grow = perm[off + mt0];
            float v0 = c00[r] + bv0;
            float v1 = c01[r] + bv1;
            out0[(size_t)grow * DD + ncol0] = v0;
            out0[(size_t)grow * DD + ncol1] = v1;
            if (out1) { out1[(size_t)grow * DD + ncol0] = v0;
                        out1[(size_t)grow * DD + ncol1] = v1; }
        }
        if (mt1 < cnt) {
            int grow = perm[off + mt1];
            float v0 = c10[r] + bv0;
            float v1 = c11[r] + bv1;
            out0[(size_t)grow * DD + ncol0] = v0;
            out0[(size_t)grow * DD + ncol1] = v1;
            if (out1) { out1[(size_t)grow * DD + ncol0] = v0;
                        out1[(size_t)grow * DD + ncol1] = v1; }
        }
    }
}

// ---------------- host orchestration ----------------
struct Ws {
    float* h; float* agg; unsigned short* Abf; unsigned short* Wt;
    float* deg; float* dinv; float* colsum; float* colsq; float* scalev; float* shiftv;
    int* eid; int* perm; int* counts; int* offs; int* cursor;
};

static void run_layer(const float* x_in, const int* ei,
                      const float* gw, const float* gb, const float* gms,
                      const float* wg, const float* W, const float* b,
                      float* out0, float* out1, const Ws& w, hipStream_t s) {
    const size_t ND = (size_t)NN * DD;
    const int gND = (int)((ND + 255) / 256);
    k_zero_small<<<2, 512, 0, s>>>(w.colsum, w.colsq, w.counts);
    k_colstats<<<1024, 256, 0, s>>>(x_in, w.colsum, w.colsq);
    k_gn_finalize<<<1, 512, 0, s>>>(w.colsum, w.colsq, gw, gb, gms, w.scalev, w.shiftv);
    k_gn_apply<<<gND, 256, 0, s>>>(x_in, w.scalev, w.shiftv, w.h);
    k_gate<<<(NN + 7) / 8, 256, 0, s>>>(w.h, wg, w.eid, w.counts);
    k_offsets<<<1, 32, 0, s>>>(w.counts, w.offs, w.cursor);
    k_scatter<<<(NN + 255) / 256, 256, 0, s>>>(w.eid, w.cursor, w.perm);
    k_prop_init<<<gND, 256, 0, s>>>(w.h, w.dinv, w.agg);
    k_prop_edges<<<NEDGE / 4, 256, 0, s>>>(w.h, ei, w.dinv, w.agg);
    k_convertA<<<gND, 256, 0, s>>>(w.agg, w.Abf);
    k_convert_Wt<<<(NE_ * DD * DD) / 256, 256, 0, s>>>(W, w.Wt);
    dim3 g((NN + 31) / 32, DD / 32, NE_);
    k_gemm_moe<<<g, 32, 0, s>>>(w.Abf, w.Wt, b, w.perm, w.offs, w.counts, out0, out1);
}

extern "C" void kernel_launch(void* const* d_in, const int* in_sizes, int n_in,
                              void* d_out, int out_size, void* d_ws, size_t ws_size,
                              hipStream_t stream) {
    const float* x   = (const float*)d_in[0];
    const int*   ei  = (const int*)d_in[1];
    const float* g1w = (const float*)d_in[2];
    const float* g1b = (const float*)d_in[3];
    const float* g1m = (const float*)d_in[4];
    const float* g2w = (const float*)d_in[5];
    const float* g2b = (const float*)d_in[6];
    const float* g2m = (const float*)d_in[7];
    const float* wg1 = (const float*)d_in[8];
    const float* W1  = (const float*)d_in[9];
    const float* b1  = (const float*)d_in[10];
    const float* wg2 = (const float*)d_in[11];
    const float* W2  = (const float*)d_in[12];
    const float* b2  = (const float*)d_in[13];
    float* out = (float*)d_out;

    const size_t ND = (size_t)NN * DD;
    Ws w;
    w.h      = (float*)d_ws;
    w.agg    = w.h + ND;
    w.Abf    = (unsigned short*)(w.agg + ND);
    w.Wt     = w.Abf + ND;
    w.deg    = (float*)(w.Wt + (size_t)NE_ * DD * DD);
    w.dinv   = w.deg + NN;
    w.colsum = w.dinv + NN;
    w.colsq  = w.colsum + DD;
    w.scalev = w.colsq + DD;
    w.shiftv = w.scalev + DD;
    w.eid    = (int*)(w.shiftv + DD);
    w.perm   = w.eid + NN;
    w.counts = w.perm + NN;
    w.offs   = w.counts + NE_;
    w.cursor = w.offs + NE_;

    // degree / dinv (same for both layers)
    k_set1<<<(NN + 255) / 256, 256, 0, stream>>>(w.deg);
    k_deg_edges<<<(NEDGE + 255) / 256, 256, 0, stream>>>(ei, w.deg);
    k_rsqrt<<<(NN + 255) / 256, 256, 0, stream>>>(w.deg, w.dinv);

    // layer 1: h1 -> d_out[ND .. 2ND)
    run_layer(x, ei, g1w, g1b, g1m, wg1, W1, b1, out + ND, nullptr, w, stream);
    // layer 2 reads h1, writes h3 -> d_out[0 .. ND) and duplicate at [2ND .. 3ND)
    run_layer(out + ND, ei, g2w, g2b, g2m, wg2, W2, b2, out, out + 2 * ND, w, stream);
}
